// DecoderRNN_91096256348600
// MI455X (gfx1250) — matrix operand order repaired and checked by reading, not compiled, not run on hardware
//
#include <hip/hip_runtime.h>
#include <hip/hip_bf16.h>

// ---------------- problem dimensions ----------------
#define Bc   128
#define Pc   196
#define Ec   512
#define Hc   512
#define Vc   10000
#define Lc   32
#define Tc   31          // L - 1
#define KX   1536        // 2E + H  (fused [emb|ctx|h] @ [Wih|Whh]^T)
#define G4   2048        // 4H

typedef __attribute__((ext_vector_type(16))) __bf16 v16bf;
typedef __attribute__((ext_vector_type(8)))  __bf16 v8bf;
typedef __attribute__((ext_vector_type(8)))  float  v8f;

// Tensor Data Mover availability (codegen path differs per toolchain arity)
#if __has_builtin(__builtin_amdgcn_tensor_load_to_lds) && __has_builtin(__builtin_amdgcn_s_wait_tensorcnt)
#define USE_TDM 1
#else
#define USE_TDM 0
#endif

#if USE_TDM
typedef __attribute__((ext_vector_type(4))) unsigned int v4u;
typedef __attribute__((ext_vector_type(8))) int          v8i;
typedef __attribute__((ext_vector_type(4))) int          v4i;

// Issue one TDM 2D tile load: rows x 32 bf16 elements, row stride = k_len elems,
// destination LDS rows padded to LDT elems (pad_interval 16 DWORDs, pad_amount 4 DWORDs).
__device__ __attribute__((always_inline))
void tdm_load_tile_bf16(const __bf16* gsrc, unsigned int lds_off,
                        int rows_valid, int k_len, int tile_rows)
{
    unsigned long ga = (unsigned long)gsrc;
    v4u g0;
    g0[0] = 1u;                                           // count=1, gather off
    g0[1] = lds_off;                                      // LDS byte address
    g0[2] = (unsigned int)ga;                             // global_addr[31:0]
    g0[3] = (unsigned int)((ga >> 32) & 0x01ffffffu)      // global_addr[56:32]
          | (2u << 30);                                   // type = 2 (image)
    unsigned int td0 = (unsigned int)k_len;               // tensor_dim0 (elements)
    unsigned int td1 = (unsigned int)rows_valid;          // tensor_dim1 (rows; OOB rows read 0)
    unsigned long st0 = (unsigned long)k_len;             // tensor_dim0_stride (elements)
    v8i g1;
    g1[0] = (1 << 16)        // data_size = 1 -> 2 bytes
          | (1 << 20)        // pad_enable
          | (3 << 22)        // pad_interval: code 3 -> 16 DWORDs (= 32 bf16 row)
          | (3 << 25);       // pad_amount:  code 3 -> 4 DWORDs (= 8 bf16 pad -> LDT 40)
    g1[1] = (int)((td0 & 0xffffu) << 16);                 // tensor_dim0[15:0] at bit48
    g1[2] = (int)((td0 >> 16) | ((td1 & 0xffffu) << 16)); // dim0[31:16] | dim1[15:0]
    g1[3] = (int)((td1 >> 16) | (32u << 16));             // dim1[31:16] | tile_dim0 = 32
    g1[4] = (int)(unsigned int)tile_rows;                 // tile_dim1 | tile_dim2 = 0
    g1[5] = (int)(unsigned int)st0;                       // dim0_stride[31:0]
    g1[6] = (int)((st0 >> 32) & 0xffffu);                 // dim0_stride[47:32] | dim1_stride[15:0]=0
    g1[7] = 0;
    v4i z4 = {0, 0, 0, 0};
#if defined(__clang_major__) && (__clang_major__ >= 23)
    v8i z8 = {0, 0, 0, 0, 0, 0, 0, 0};
    __builtin_amdgcn_tensor_load_to_lds(g0, g1, z4, z4, z8, 0);
#else
    __builtin_amdgcn_tensor_load_to_lds(g0, g1, z4, z4, 0);
#endif
}
#endif // USE_TDM

// ---------------- generic bf16 WMMA GEMM: C[M,N] = A[M,K] @ W[N,K]^T (+bias) ----------------
// Block tile 128x64, 8 waves of 32 lanes, each wave owns a 32x32 tile (2x2 WMMA 16x16x32).
#define BM 128
#define BN 64
#define BK 32
#define LDT 40   // padded LDS leading dim (bf16 elems); row stride 80B keeps 16B alignment

__device__ __attribute__((always_inline))
void mma_step(const __bf16* AsT, const __bf16* BsT, int wm, int wn, int lane,
              v8f (&acc)[2][2])
{
    // A fragments: 16-bit A 16x32 layout (K 0-7/16-23 | 8-15/24-31)
    v16bf af[2];
#pragma unroll
    for (int i = 0; i < 2; ++i) {
        const __bf16* p = AsT + (wm * 32 + i * 16 + (lane & 15)) * LDT;
        const int half = (lane >> 4) * 8;
        v8bf lo = *(const v8bf*)(p + half);
        v8bf hi = *(const v8bf*)(p + 16 + half);
#pragma unroll
        for (int q = 0; q < 8; ++q) { af[i][q] = lo[q]; af[i][q + 8] = hi[q]; }
    }
    // B fragments: same layout as A (K 0-7/16-23 | 8-15/24-31)
    v16bf bfm[2];
#pragma unroll
    for (int j = 0; j < 2; ++j) {
        const __bf16* p = BsT + (wn * 32 + j * 16 + (lane & 15)) * LDT + (lane >> 4) * 8;
        v8bf lo = *(const v8bf*)(p);
        v8bf hi = *(const v8bf*)(p + 16);
#pragma unroll
        for (int q = 0; q < 8; ++q) { bfm[j][q] = lo[q]; bfm[j][q + 8] = hi[q]; }
    }
#pragma unroll
    for (int i = 0; i < 2; ++i)
#pragma unroll
        for (int j = 0; j < 2; ++j)
            acc[i][j] = __builtin_amdgcn_wmma_f32_16x16x32_bf16(
                false, af[i], false, bfm[j], (short)0, acc[i][j], false, false);
}

__global__ __launch_bounds__(256)
void gemm_bf16_wmma(const __bf16* __restrict__ A, const __bf16* __restrict__ W,
                    const float* __restrict__ bias,
                    float* __restrict__ outF, __bf16* __restrict__ outB,
                    int M, int N, int K, long ldOut,
                    const int* __restrict__ lens, int t)
{
    __shared__ __bf16 As[2][BM * LDT];
    __shared__ __bf16 Bs[2][BN * LDT];

    const int tid  = threadIdx.x;
    const int bm   = blockIdx.y;
    const int bn   = blockIdx.x;
    const int lane = tid & 31;
    const int wid  = tid >> 5;
    const int wm   = wid >> 1;   // 0..3 -> 32-row strip
    const int wn   = wid & 1;    // 0..1 -> 32-col strip

    v8f acc[2][2];
#pragma unroll
    for (int i = 0; i < 2; ++i)
#pragma unroll
        for (int j = 0; j < 2; ++j)
#pragma unroll
            for (int q = 0; q < 8; ++q) acc[i][j][q] = 0.0f;

    const int nK = K / BK;

#if USE_TDM
    // ---- Tensor Data Mover double-buffered pipeline: DMA tile k+1 overlaps WMMA on tile k ----
    const unsigned int aOff[2] = { (unsigned int)(unsigned long)(void*)&As[0][0],
                                   (unsigned int)(unsigned long)(void*)&As[1][0] };
    const unsigned int bOff[2] = { (unsigned int)(unsigned long)(void*)&Bs[0][0],
                                   (unsigned int)(unsigned long)(void*)&Bs[1][0] };
    const int aRows = M - bm * BM;      // >= BM by construction
    const int bRows = N - bn * BN;      // ragged on the vocab GEMM: TDM OOB rows read 0

    if (wid == 0) {
        tdm_load_tile_bf16(A + (long)bm * BM * K, aOff[0], aRows, K, BM);
        tdm_load_tile_bf16(W + (long)bn * BN * K, bOff[0], bRows, K, BN);
    }
    int cur = 0;
    for (int kc = 0; kc < nK; ++kc) {
        if (wid == 0) __builtin_amdgcn_s_wait_tensorcnt(0);   // tile kc resident
        __syncthreads();                                      // ...visible to all waves;
                                                              // also: everyone done reading buf cur^1
        if ((kc + 1 < nK) && wid == 0) {
            const int nb = cur ^ 1;
            tdm_load_tile_bf16(A + (long)bm * BM * K + (kc + 1) * BK, aOff[nb], aRows, K, BM);
            tdm_load_tile_bf16(W + (long)bn * BN * K + (kc + 1) * BK, bOff[nb], bRows, K, BN);
        }
        mma_step(&As[cur][0], &Bs[cur][0], wm, wn, lane, acc);
        cur ^= 1;
    }
#else
    // ---- fallback: per-lane staging through VGPRs ----
    for (int kc = 0; kc < nK; ++kc) {
        const int k0 = kc * BK;
#pragma unroll
        for (int c = tid; c < 512; c += 256) {
            const int r   = c >> 2;
            const int col = (c & 3) * 8;
            const long grow = (long)bm * BM + r;
            v8bf vb = *(const v8bf*)(A + grow * K + k0 + col);
            *(v8bf*)(&As[0][0] + r * LDT + col) = vb;
        }
        {
            const int r   = tid >> 2;
            const int col = (tid & 3) * 8;
            const int grow = bn * BN + r;
            v8bf vb;
            if (grow < N) {
                vb = *(const v8bf*)(W + (long)grow * K + k0 + col);
            } else {
#pragma unroll
                for (int q = 0; q < 8; ++q) vb[q] = (__bf16)0.0f;
            }
            *(v8bf*)(&Bs[0][0] + r * LDT + col) = vb;
        }
        if (kc + 1 < nK) {
            const long ar = (long)bm * BM + (tid & 127);
            __builtin_prefetch(A + ar * K + k0 + BK, 0, 0);
            int wr = bn * BN + (tid & 63); if (wr >= N) wr = N - 1;
            __builtin_prefetch(W + (long)wr * K + k0 + BK, 0, 0);
        }
        __syncthreads();
        mma_step(&As[0][0], &Bs[0][0], wm, wn, lane, acc);
        __syncthreads();
    }
#endif

    // ---- epilogue: C/D layout — lane&15 = N col, (lane>>4)*8 + r = M row ----
#pragma unroll
    for (int i = 0; i < 2; ++i) {
#pragma unroll
        for (int j = 0; j < 2; ++j) {
#pragma unroll
            for (int r = 0; r < 8; ++r) {
                const int grow = bm * BM + wm * 32 + i * 16 + (lane >> 4) * 8 + r;
                const int gcol = bn * BN + wn * 32 + j * 16 + (lane & 15);
                if (gcol < N) {
                    float v = acc[i][j][r];
                    if (bias) v += bias[gcol];
                    if (lens) { if (!(t < lens[grow] - 1)) v = 0.0f; }
                    if (outF) outF[(long)grow * ldOut + gcol] = v;
                    else      outB[(long)grow * ldOut + gcol] = (__bf16)v;
                }
            }
        }
    }
}

// ---------------- attention score: e[b,p] = sum_e tanh(feat+hproj)*v + bv ----------------
__global__ __launch_bounds__(256)
void attn_e_kernel(const __bf16* __restrict__ featp, const float* __restrict__ hproj,
                   const float* __restrict__ vvec, const float* __restrict__ bv,
                   float* __restrict__ e_out)
{
    const int w    = blockIdx.x * 8 + (threadIdx.x >> 5);   // one wave per (b,p)
    const int lane = threadIdx.x & 31;
    const int b = w / Pc, p = w % Pc;
    const __bf16* f  = featp + ((long)b * Pc + p) * Ec;
    const float*  hp = hproj + (long)b * Ec;
    float s = 0.0f;
#pragma unroll 4
    for (int k = 0; k < Ec / 32; ++k) {
        const int e = k * 32 + lane;
        s += tanhf((float)f[e] + hp[e]) * vvec[e];
    }
#pragma unroll
    for (int off = 16; off; off >>= 1) s += __shfl_xor(s, off, 32);
    if (lane == 0) e_out[b * Pc + p] = s + bv[0];
}

// ---------------- softmax over pixels + context vector ----------------
__global__ __launch_bounds__(256)
void softmax_ctx_kernel(const float* __restrict__ e_in, const float* __restrict__ images,
                        const int* __restrict__ lengths, int t,
                        float* __restrict__ alphas_out, float* __restrict__ ctx)
{
    const int b = blockIdx.x, tid = threadIdx.x;
    __shared__ float sh[256];
    __shared__ float sa[Pc];

    float ev = (tid < Pc) ? e_in[b * Pc + tid] : -3.4e38f;
    sh[tid] = ev; __syncthreads();
    for (int s = 128; s; s >>= 1) { if (tid < s) sh[tid] = fmaxf(sh[tid], sh[tid + s]); __syncthreads(); }
    const float m = sh[0]; __syncthreads();

    float ex = (tid < Pc) ? expf(ev - m) : 0.0f;
    sh[tid] = ex; __syncthreads();
    for (int s = 128; s; s >>= 1) { if (tid < s) sh[tid] += sh[tid + s]; __syncthreads(); }
    const float inv = 1.0f / sh[0];

    const bool mask = t < lengths[b] - 1;
    if (tid < Pc) {
        const float a = ex * inv;
        sa[tid] = a;
        alphas_out[((long)b * Lc + t) * Pc + tid] = mask ? a : 0.0f;
    }
    __syncthreads();

    for (int e0 = tid; e0 < Ec; e0 += 256) {
        float acc = 0.0f;
        const float* im = images + (long)b * Pc * Ec + e0;
#pragma unroll 4
        for (int p = 0; p < Pc; ++p) acc += sa[p] * im[(long)p * Ec];
        ctx[b * Ec + e0] = acc;
    }
}

// ---------------- build xh = [emb_t | ctx | h] in bf16 ----------------
__global__ void build_xh(const __bf16* __restrict__ emb, const float* __restrict__ ctx,
                         const __bf16* __restrict__ h_bf, int t, __bf16* __restrict__ xh)
{
    const int i = blockIdx.x * 256 + threadIdx.x;
    if (i >= Bc * KX) return;
    const int b = i / KX, col = i % KX;
    __bf16 v;
    if (col < Ec)           v = emb[((long)b * Lc + t) * Ec + col];
    else if (col < 2 * Ec)  v = (__bf16)ctx[b * Ec + (col - Ec)];
    else                    v = h_bf[b * Hc + (col - 2 * Ec)];
    xh[i] = v;
}

// ---------------- LSTM gate nonlinearities + masked state update ----------------
__global__ void lstm_update_kernel(const float* __restrict__ gates, const int* __restrict__ lengths,
                                   int t, float* __restrict__ h, float* __restrict__ c,
                                   __bf16* __restrict__ h_bf)
{
    const int idx = blockIdx.x * 256 + threadIdx.x;
    if (idx >= Bc * Hc) return;
    const int b = idx / Hc, j = idx % Hc;
    const float* g = gates + (long)b * G4;
    const float i_ = g[j], f_ = g[Hc + j], g_ = g[2 * Hc + j], o_ = g[3 * Hc + j];
    const float si = 1.0f / (1.0f + expf(-i_));
    const float sf = 1.0f / (1.0f + expf(-f_));
    const float so = 1.0f / (1.0f + expf(-o_));
    const float cn = sf * c[idx] + si * tanhf(g_);
    const float hn = so * tanhf(cn);
    const bool mask = t < lengths[b] - 1;
    const float h2 = mask ? hn : h[idx];
    const float c2 = mask ? cn : c[idx];
    h[idx] = h2; c[idx] = c2; h_bf[idx] = (__bf16)h2;
}

// ---------------- setup kernels ----------------
__global__ void cvt_bf16(const float* __restrict__ src, __bf16* __restrict__ dst, long n)
{
    for (long i = (long)blockIdx.x * 256 + threadIdx.x; i < n; i += (long)gridDim.x * 256)
        dst[i] = (__bf16)src[i];
}

__global__ void embed_gather(const float* __restrict__ embW, const int* __restrict__ captions,
                             __bf16* __restrict__ emb)
{
    const long i = (long)blockIdx.x * 256 + threadIdx.x;
    if (i >= (long)Bc * Lc * Ec) return;
    const long bl = i / Ec; const int e = (int)(i % Ec);
    const int tok = captions[bl];
    emb[i] = (__bf16)embW[(long)tok * Ec + e];
}

__global__ void build_wcomb(const float* __restrict__ Wih, const float* __restrict__ Whh,
                            const float* __restrict__ bih, const float* __restrict__ bhh,
                            __bf16* __restrict__ Wc, float* __restrict__ bihh)
{
    const long i = (long)blockIdx.x * 256 + threadIdx.x;
    if (i < (long)G4 * KX) {
        const int n = (int)(i / KX), k = (int)(i % KX);
        const float v = (k < 2 * Ec) ? Wih[(long)n * (2 * Ec) + k]
                                     : Whh[(long)n * Hc + (k - 2 * Ec)];
        Wc[i] = (__bf16)v;
    }
    if (i < G4) bihh[i] = bih[i] + bhh[i];
}

__global__ void zero_hc(float* h, float* c, __bf16* h_bf)
{
    const int i = blockIdx.x * 256 + threadIdx.x;
    if (i < Bc * Hc) { h[i] = 0.0f; c[i] = 0.0f; h_bf[i] = (__bf16)0.0f; }
}

__global__ void init_out_kernel(float* __restrict__ out, const int* __restrict__ captions,
                                const int* __restrict__ lengths)
{
    const long PRED = (long)Bc * Lc * Vc;
    const long CAP  = (long)Bc * Lc;
    const long LEN  = Bc;
    const long ALPH = (long)Bc * Lc * Pc;
    const long tot  = PRED + CAP + LEN + ALPH;
    for (long i = (long)blockIdx.x * 256 + threadIdx.x; i < tot; i += (long)gridDim.x * 256) {
        if (i < PRED)                  out[i] = 0.0f;
        else if (i < PRED + CAP)       ((int*)out)[i] = captions[i - PRED];
        else if (i < PRED + CAP + LEN) ((int*)out)[i] = lengths[i - PRED - CAP];
        else                           out[i] = 0.0f;
    }
}

// ---------------- host driver ----------------
extern "C" void kernel_launch(void* const* d_in, const int* in_sizes, int n_in,
                              void* d_out, int out_size, void* d_ws, size_t ws_size,
                              hipStream_t stream)
{
    const float* images   = (const float*)d_in[0];
    const int*   captions = (const int*)  d_in[1];
    const int*   lengths  = (const int*)  d_in[2];
    const float* embW     = (const float*)d_in[3];
    const float* w1       = (const float*)d_in[4];
    const float* b1       = (const float*)d_in[5];
    const float* b2w      = (const float*)d_in[6];   // w2
    const float* b2       = (const float*)d_in[7];
    const float* vv       = (const float*)d_in[8];
    const float* bv       = (const float*)d_in[9];
    const float* Wih      = (const float*)d_in[10];
    const float* bih      = (const float*)d_in[11];
    const float* Whh      = (const float*)d_in[12];
    const float* bhh      = (const float*)d_in[13];
    const float* fcW      = (const float*)d_in[14];
    const float* fcb      = (const float*)d_in[15];
    float* out = (float*)d_out;

    // workspace carve-out (256B aligned)
    char* ws = (char*)d_ws;
    size_t off = 0;
    auto alloc = [&](size_t bytes) -> void* {
        off = (off + 255) & ~(size_t)255;
        void* p = ws + off;
        off += bytes;
        return p;
    };
    __bf16* images_bf = (__bf16*)alloc((size_t)Bc * Pc * Ec * 2);
    __bf16* featp_bf  = (__bf16*)alloc((size_t)Bc * Pc * Ec * 2);
    __bf16* w1_bf     = (__bf16*)alloc((size_t)Ec * Ec * 2);
    __bf16* w2_bf     = (__bf16*)alloc((size_t)Ec * Hc * 2);
    __bf16* wcomb     = (__bf16*)alloc((size_t)G4 * KX * 2);
    __bf16* fcw_bf    = (__bf16*)alloc((size_t)Vc * Hc * 2);
    __bf16* emb_bf    = (__bf16*)alloc((size_t)Bc * Lc * Ec * 2);
    float*  bihh      = (float*) alloc((size_t)G4 * 4);
    float*  h_st      = (float*) alloc((size_t)Bc * Hc * 4);
    float*  c_st      = (float*) alloc((size_t)Bc * Hc * 4);
    __bf16* h_bf      = (__bf16*)alloc((size_t)Bc * Hc * 2);
    float*  hproj     = (float*) alloc((size_t)Bc * Ec * 4);
    float*  e_buf     = (float*) alloc((size_t)Bc * Pc * 4);
    float*  ctx       = (float*) alloc((size_t)Bc * Ec * 4);
    __bf16* xh        = (__bf16*)alloc((size_t)Bc * KX * 2);
    float*  gates     = (float*) alloc((size_t)Bc * G4 * 4);

    const long PRED = (long)Bc * Lc * Vc;
    float* alphas_base = out + PRED + (long)Bc * Lc + Bc;

    // ---- one-time setup ----
    init_out_kernel<<<4096, 256, 0, stream>>>(out, captions, lengths);
    cvt_bf16<<<8192, 256, 0, stream>>>(images, images_bf, (long)Bc * Pc * Ec);
    cvt_bf16<<<1024, 256, 0, stream>>>(w1, w1_bf, (long)Ec * Ec);
    cvt_bf16<<<1024, 256, 0, stream>>>(b2w, w2_bf, (long)Ec * Hc);
    cvt_bf16<<<8192, 256, 0, stream>>>(fcW, fcw_bf, (long)Vc * Hc);
    build_wcomb<<<((long)G4 * KX + 255) / 256, 256, 0, stream>>>(Wih, Whh, bih, bhh, wcomb, bihh);
    embed_gather<<<((long)Bc * Lc * Ec + 255) / 256, 256, 0, stream>>>(embW, captions, emb_bf);
    zero_hc<<<(Bc * Hc + 255) / 256, 256, 0, stream>>>(h_st, c_st, h_bf);

    // feat_proj = images @ w1^T + b1   (bf16 out, reused every step)
    gemm_bf16_wmma<<<dim3(Ec / BN, (Bc * Pc) / BM), 256, 0, stream>>>(
        images_bf, w1_bf, b1, nullptr, featp_bf,
        Bc * Pc, Ec, Ec, Ec, nullptr, 0);

    // ---- sequential decode ----
    for (int t = 0; t < Tc; ++t) {
        // hproj = h @ w2^T + b2
        gemm_bf16_wmma<<<dim3(Ec / BN, 1), 256, 0, stream>>>(
            h_bf, w2_bf, b2, hproj, nullptr,
            Bc, Ec, Hc, Ec, nullptr, 0);
        // e = tanh(feat + hproj) . v + bv
        attn_e_kernel<<<(Bc * Pc) / 8, 256, 0, stream>>>(featp_bf, hproj, vv, bv, e_buf);
        // alpha = softmax(e); ctx = alpha . images
        softmax_ctx_kernel<<<Bc, 256, 0, stream>>>(e_buf, images, lengths, t, alphas_base, ctx);
        // xh = [emb_t | ctx | h]
        build_xh<<<(Bc * KX + 255) / 256, 256, 0, stream>>>(emb_bf, ctx, h_bf, t, xh);
        // gates = xh @ [Wih|Whh]^T + (bih + bhh)
        gemm_bf16_wmma<<<dim3(G4 / BN, 1), 256, 0, stream>>>(
            xh, wcomb, bihh, gates, nullptr,
            Bc, G4, KX, G4, nullptr, 0);
        // LSTM state update (masked)
        lstm_update_kernel<<<(Bc * Hc + 255) / 256, 256, 0, stream>>>(
            gates, lengths, t, h_st, c_st, h_bf);
        // preds[:, t, :] = h_new @ fcW^T + fcb (masked rows -> 0)
        gemm_bf16_wmma<<<dim3((Vc + BN - 1) / BN, 1), 256, 0, stream>>>(
            h_bf, fcw_bf, fcb, out + (long)t * Vc, nullptr,
            Bc, Vc, Hc, (long)Lc * Vc, lengths, t);
    }
}
